// MedicalEncoder_541165879646
// MI455X (gfx1250) — matrix operand order repaired and checked by reading, not compiled, not run on hardware
//
#include <hip/hip_runtime.h>
#include <math.h>

typedef __attribute__((ext_vector_type(16))) __bf16 v16bf;
typedef __attribute__((ext_vector_type(8)))  __bf16 v8bf;
typedef __attribute__((ext_vector_type(8)))  float  v8f;

#define DMODEL 384
#define NHEAD  6
#define ASTRIDE 392   // 64-row A tile stride in bf16 (16B aligned, bank-spread)
#define ASTRIDE_R 72  // reltrans A tile stride

// ---------------------------------------------------------------------------
// Encoder: masked mean pool over 16 tokens, then L2 normalize. 1 block / node.
// ---------------------------------------------------------------------------
__global__ __launch_bounds__(128)
void encode_kernel(const float* __restrict__ X, float* __restrict__ Enc)
{
    __shared__ float s[DMODEL];
    __shared__ float red[128];
    const int n = blockIdx.x, tid = threadIdx.x;
    const float* xp = X + (size_t)n * 16 * DMODEL;
    float sq = 0.f;
    for (int d = tid; d < DMODEL; d += 128) {
        float sum = 0.f, cnt = 0.f;
#pragma unroll
        for (int t = 0; t < 16; t++) {
            float v = xp[t * DMODEL + d];
            if (v > 0.f) { sum += v; cnt += 1.f; }
        }
        float val = sum / fmaxf(cnt, 1e-9f);
        s[d] = val;
        sq += val * val;
    }
    red[tid] = sq;
    __syncthreads();
    for (int stp = 64; stp > 0; stp >>= 1) {
        if (tid < stp) red[tid] += red[tid + stp];
        __syncthreads();
    }
    const float norm = fmaxf(sqrtf(red[0]), 1e-12f);
    for (int d = tid; d < DMODEL; d += 128)
        Enc[(size_t)n * DMODEL + d] = s[d] / norm;
}

// ---------------------------------------------------------------------------
// Repack f32 row-major W[K][N] (+batch) into bf16 WMMA-B tiles:
//   Wp[((ntile*(K/32) + kstep)*32 + lane)*16 + j] = W[kstep*32 + koff + j][ntile*16 + nl]
// with nl = lane&15, koff = (lane>=16)*16.  One thread per 16 bf16 outputs.
// ---------------------------------------------------------------------------
__global__ void repack_b(const float* __restrict__ W, __bf16* __restrict__ Wp,
                         int K, int N)
{
    const int nk = K / 32, nn = N / 16;
    const int total = nn * nk * 32;
    int gid = blockIdx.x * 256 + threadIdx.x;
    if (gid >= total) return;
    const float* Wb = W + (size_t)blockIdx.y * K * N;
    __bf16* Wpb = Wp + (size_t)blockIdx.y * K * N;   // K*N halfs per batch
    const int lane = gid & 31, rest = gid >> 5;
    const int kstep = rest % nk, ntile = rest / nk;
    const int nl = lane & 15, koff = (lane >> 4) * 16;
    const float* src = Wb + (size_t)(kstep * 32 + koff) * N + ntile * 16 + nl;
    __bf16* dst = Wpb + (size_t)gid * 16;
#pragma unroll
    for (int j = 0; j < 16; j++) dst[j] = (__bf16)src[(size_t)j * N];
}

// ---------------------------------------------------------------------------
// 384x384 GEMM via v_wmma_f32_16x16x32_bf16.
// Block = 4 waves; block tile 64(M) x 64(N); each wave: 16 N-cols x 4 M-tiles.
// A staged in LDS (bf16, padded stride); B from repacked bf16 (1 load / wmma grp).
// MODE: 0 = +bias, 1 = +bias,relu, 2 = +bias then skip-blend with xold.
// ---------------------------------------------------------------------------
template<int MODE>
__global__ __launch_bounds__(128)
void gemm384(const float* __restrict__ X, const __bf16* __restrict__ Wp,
             const float* __restrict__ Bv, float* __restrict__ Y,
             const float* __restrict__ skipp, const float* __restrict__ xold)
{
    __shared__ __bf16 As[64 * ASTRIDE];
    const int m0  = blockIdx.x * 64;
    const int tid = threadIdx.x;
    for (int r = 0; r < 64; r++) {
        const float* xr = X + (size_t)(m0 + r) * DMODEL;
        __bf16* ar = As + r * ASTRIDE;
        for (int c = tid; c < DMODEL; c += 128) ar[c] = (__bf16)xr[c];
    }
    __syncthreads();

    const int wave = tid >> 5, lane = tid & 31;
    const int hi = lane >> 4, nl = lane & 15;
    const int ncolTile = blockIdx.y * 4 + wave;
    const int ncol = ncolTile * 16 + nl;

    v8f acc[4];
#pragma unroll
    for (int mt = 0; mt < 4; mt++) acc[mt] = (v8f){0.f,0.f,0.f,0.f,0.f,0.f,0.f,0.f};

    for (int kb12 = 0; kb12 < 12; kb12++) {
        const int kb = kb12 * 32;
        const v16bf b = *(const v16bf*)(Wp + (size_t)((ncolTile * 12 + kb12) * 32 + lane) * 16);
#pragma unroll
        for (int mt = 0; mt < 4; mt++) {
            const __bf16* arow = &As[(mt * 16 + nl) * ASTRIDE + kb];
            v8bf lo = *(const v8bf*)(arow + (hi ? 16 : 0));
            v8bf hi8 = *(const v8bf*)(arow + (hi ? 24 : 8));
            v16bf a = __builtin_shufflevector(lo, hi8, 0,1,2,3,4,5,6,7,8,9,10,11,12,13,14,15);
            acc[mt] = __builtin_amdgcn_wmma_f32_16x16x32_bf16(false, a, false, b,
                                                              (short)0, acc[mt], false, false);
        }
    }

    const float bias = Bv[ncol];
    float sk = 0.f;
    if (MODE == 2) sk = 1.f / (1.f + expf(-skipp[0]));
#pragma unroll
    for (int mt = 0; mt < 4; mt++) {
#pragma unroll
        for (int r = 0; r < 8; r++) {
            const int m = m0 + mt * 16 + r + (hi ? 8 : 0);
            float o = acc[mt][r] + bias;
            if (MODE == 1) o = fmaxf(o, 0.f);
            if (MODE == 2) o = sk * o + (1.f - sk) * xold[(size_t)m * DMODEL + ncol];
            Y[(size_t)m * DMODEL + ncol] = o;
        }
    }
}

// ---------------------------------------------------------------------------
// Relation transform: out[n,h,:] = in[n,h,:] @ rel[h] (64x64 / head), WMMA.
// grid = (Ns/64, 6 heads, 2: 0 -> KR = K @ a_rel, 1 -> VR = V @ m_rel).
// Block tile 64(M) x 64(e); rel repacked bf16; A tile (64x64) in LDS.
// ---------------------------------------------------------------------------
__global__ __launch_bounds__(128)
void reltrans(const float* __restrict__ Kin, const float* __restrict__ Vin,
              const __bf16* __restrict__ ArelP, const __bf16* __restrict__ MrelP,
              float* __restrict__ KR, float* __restrict__ VR)
{
    const int h = blockIdx.y;
    const float*  Xin = blockIdx.z ? Vin   : Kin;
    const __bf16* Rp  = blockIdx.z ? MrelP : ArelP;
    float*        Out = blockIdx.z ? VR    : KR;

    __shared__ __bf16 As[64 * ASTRIDE_R];
    const int m0 = blockIdx.x * 64;
    const int tid = threadIdx.x;
    for (int idx = tid; idx < 4096; idx += 128) {
        const int row = idx >> 6, col = idx & 63;
        As[row * ASTRIDE_R + col] = (__bf16)Xin[(size_t)(m0 + row) * DMODEL + h * 64 + col];
    }
    __syncthreads();

    const int wave = tid >> 5, lane = tid & 31;
    const int hi = lane >> 4, nl = lane & 15;
    const int etile = wave;                       // 4 waves cover 64 outputs

    v8f acc[4];
#pragma unroll
    for (int mt = 0; mt < 4; mt++) acc[mt] = (v8f){0.f,0.f,0.f,0.f,0.f,0.f,0.f,0.f};

#pragma unroll
    for (int ks = 0; ks < 2; ks++) {
        const int kb = ks * 32;
        const v16bf b = *(const v16bf*)(Rp + (size_t)h * 4096 +
                                        (size_t)((etile * 2 + ks) * 32 + lane) * 16);
#pragma unroll
        for (int mt = 0; mt < 4; mt++) {
            const __bf16* arow = &As[(mt * 16 + nl) * ASTRIDE_R + kb];
            v8bf lo = *(const v8bf*)(arow + (hi ? 16 : 0));
            v8bf hi8 = *(const v8bf*)(arow + (hi ? 24 : 8));
            v16bf a = __builtin_shufflevector(lo, hi8, 0,1,2,3,4,5,6,7,8,9,10,11,12,13,14,15);
            acc[mt] = __builtin_amdgcn_wmma_f32_16x16x32_bf16(false, a, false, b,
                                                              (short)0, acc[mt], false, false);
        }
    }
#pragma unroll
    for (int mt = 0; mt < 4; mt++)
#pragma unroll
        for (int r = 0; r < 8; r++)
            Out[(size_t)(m0 + mt * 16 + r + (hi ? 8 : 0)) * DMODEL + h * 64 + etile * 16 + nl]
                = acc[mt][r];
}

// ---------------------------------------------------------------------------
// Edge-phase kernels (f32, atomics).
// ---------------------------------------------------------------------------
__device__ __forceinline__ unsigned fenc(float f) {
    unsigned u = __float_as_uint(f);
    return (u & 0x80000000u) ? ~u : (u | 0x80000000u);
}
__device__ __forceinline__ float fdec(unsigned u) {
    return (u & 0x80000000u) ? __uint_as_float(u ^ 0x80000000u) : __uint_as_float(~u);
}

__global__ void init_softmax(unsigned* __restrict__ amax, float* __restrict__ denom, int n)
{
    int i = blockIdx.x * 256 + threadIdx.x;
    if (i < n) { amax[i] = 0x007FFFFFu; denom[i] = 0.f; }   // enc(-inf)
}

__global__ void alpha_kernel(const float* __restrict__ Q, const float* __restrict__ KR,
                             const int* __restrict__ ei, const float* __restrict__ prel,
                             float* __restrict__ alpha, unsigned* __restrict__ amax, int E)
{
    int gid = blockIdx.x * 256 + threadIdx.x;
    if (gid >= E * NHEAD) return;
    const int e = gid / NHEAD, h = gid % NHEAD;
    const int src = ei[e], dst = ei[E + e];
    const float4* qp = (const float4*)(Q  + (size_t)dst * DMODEL + h * 64);
    const float4* kp = (const float4*)(KR + (size_t)src * DMODEL + h * 64);
    float acc = 0.f;
#pragma unroll
    for (int i = 0; i < 16; i++) {
        float4 qa = qp[i], kb = kp[i];
        acc += qa.x * kb.x + qa.y * kb.y + qa.z * kb.z + qa.w * kb.w;
    }
    const float al = acc * prel[h] * 0.125f;     // / sqrt(64)
    alpha[gid] = al;
    atomicMax(&amax[dst * NHEAD + h], fenc(al));
}

__global__ void expdenom_kernel(const int* __restrict__ ei, float* __restrict__ alpha,
                                const unsigned* __restrict__ amax,
                                float* __restrict__ denom, int E)
{
    int gid = blockIdx.x * 256 + threadIdx.x;
    if (gid >= E * NHEAD) return;
    const int e = gid / NHEAD, h = gid % NHEAD;
    const int dst = ei[E + e];
    float m = fdec(amax[dst * NHEAD + h]);
    if (!isfinite(m)) m = 0.f;
    const float ev = expf(alpha[gid] - m);
    alpha[gid] = ev;
    atomicAdd(&denom[dst * NHEAD + h], ev);
}

__global__ void scatter_kernel(const int* __restrict__ ei, const float* __restrict__ alpha,
                               const float* __restrict__ denom, const float* __restrict__ VR,
                               float* __restrict__ Out, int E)
{
    int gid = blockIdx.x * 256 + threadIdx.x;
    if (gid >= E * NHEAD * 16) return;
    const int c = gid & 15, eh = gid >> 4;
    const int e = eh / NHEAD, h = eh % NHEAD;
    const int src = ei[e], dst = ei[E + e];
    const float attn = alpha[eh] / (denom[dst * NHEAD + h] + 1e-16f);
    const float4 v = ((const float4*)(VR + (size_t)src * DMODEL + h * 64))[c];
    float* o = Out + (size_t)dst * DMODEL + h * 64 + c * 4;
    atomicAdd(o + 0, v.x * attn);
    atomicAdd(o + 1, v.y * attn);
    atomicAdd(o + 2, v.z * attn);
    atomicAdd(o + 3, v.w * attn);
}

__global__ void gelu_kernel(const float* __restrict__ in, float* __restrict__ out, int n)
{
    int i = blockIdx.x * 256 + threadIdx.x;
    if (i < n) {
        float x = in[i];
        out[i] = 0.5f * x * (1.f + erff(x * 0.70710678118654752f));
    }
}

// ---------------------------------------------------------------------------
// Host orchestration
// ---------------------------------------------------------------------------
static const int NN[3] = {8000, 8000, 24000};         // answer, question, concept
struct EdgeT { int s, d, E; };
static const EdgeT ETS[6] = {
    {1, 0, 150000}, {2, 0, 300000}, {2, 1, 300000},
    {0, 2, 300000}, {1, 2, 300000}, {0, 1, 150000},
};

extern "C" void kernel_launch(void* const* d_in, const int* in_sizes, int n_in,
                              void* d_out, int out_size, void* d_ws, size_t ws_size,
                              hipStream_t stream)
{
    (void)n_in; (void)out_size; (void)ws_size;
    auto F = [&](int i) { return (const float*)d_in[i]; };

    // ---- parameter tables; handle both insertion-order and JAX-sorted flattening
    const float *lin_w[3], *lin_b[3];
    const float *pw[2][4][3], *pb[2][4][3];           // layer, proj(k,q,v,a), type
    const float *skp[2][3], *arel[2][6], *mrel[2][6], *prel[2][6];

    const bool insertion = (in_sizes[9] == DMODEL * DMODEL);
    if (insertion) {
        for (int t = 0; t < 3; t++) { lin_w[t] = F(9 + 2 * t); lin_b[t] = F(9 + 2 * t + 1); }
        for (int L = 0; L < 2; L++) {
            const int lb = 15 + L * 45;
            for (int p = 0; p < 4; p++)
                for (int t = 0; t < 3; t++) {
                    pw[L][p][t] = F(lb + p * 6 + t * 2);
                    pb[L][p][t] = F(lb + p * 6 + t * 2 + 1);
                }
            for (int t = 0; t < 3; t++) skp[L][t] = F(lb + 24 + t);
            for (int e = 0; e < 6; e++) {
                arel[L][e] = F(lb + 27 + e);
                mrel[L][e] = F(lb + 33 + e);
                prel[L][e] = F(lb + 39 + e);
            }
        }
    } else {  // sorted: layers first, keys a,a_rel,k,m_rel,p_rel,q,skip,v; types ans,con,que; b<w
        static const int sp[3] = {0, 2, 1};
        static const int se[6] = {4, 2, 3, 0, 5, 1};
        static const int po[4] = {12, 30, 39, 0};
        for (int L = 0; L < 2; L++) {
            const int lb = 9 + L * 45;
            for (int p = 0; p < 4; p++)
                for (int t = 0; t < 3; t++) {
                    pb[L][p][t] = F(lb + po[p] + 2 * sp[t]);
                    pw[L][p][t] = F(lb + po[p] + 2 * sp[t] + 1);
                }
            for (int t = 0; t < 3; t++) skp[L][t] = F(lb + 36 + sp[t]);
            for (int e = 0; e < 6; e++) {
                arel[L][e] = F(lb + 6  + se[e]);
                mrel[L][e] = F(lb + 18 + se[e]);
                prel[L][e] = F(lb + 24 + se[e]);
            }
        }
        for (int t = 0; t < 3; t++) { lin_b[t] = F(99 + 2 * sp[t]); lin_w[t] = F(99 + 2 * sp[t] + 1); }
    }

    const int* ei[6];
    for (int e = 0; e < 6; e++) ei[e] = (const int*)d_in[3 + e];

    // ---- workspace (bump allocator, 256B aligned)
    char* ws = (char*)d_ws; size_t off = 0;
    auto take = [&](size_t b) { char* p = ws + off; off += (b + 255) & ~(size_t)255; return p; };
    const size_t fb[3] = {(size_t)NN[0]*DMODEL*4, (size_t)NN[1]*DMODEL*4, (size_t)NN[2]*DMODEL*4};
    float *x[3], *xn[3], *kb[3], *qb[3], *vb[3], *ob[3];
    for (int t = 0; t < 3; t++) x[t]  = (float*)take(fb[t]);
    for (int t = 0; t < 3; t++) xn[t] = (float*)take(fb[t]);
    for (int t = 0; t < 3; t++) kb[t] = (float*)take(fb[t]);
    for (int t = 0; t < 3; t++) qb[t] = (float*)take(fb[t]);
    for (int t = 0; t < 3; t++) vb[t] = (float*)take(fb[t]);
    for (int t = 0; t < 3; t++) ob[t] = (float*)take(fb[t]);
    float*    krb    = (float*)take((size_t)24000 * DMODEL * 4);
    float*    vrb    = (float*)take((size_t)24000 * DMODEL * 4);
    float*    gbuf   = (float*)take((size_t)24000 * DMODEL * 4);
    float*    alphaB = (float*)take((size_t)300000 * NHEAD * 4);
    unsigned* amaxB  = (unsigned*)take((size_t)24000 * NHEAD * 4);
    float*    denomB = (float*)take((size_t)24000 * NHEAD * 4);
    // bf16 repacked weights
    const size_t WSZ = (size_t)DMODEL * DMODEL;          // halfs per 384x384
    __bf16 *wp_lin[3], *wp_proj[2][4][3], *rp_a[2][6], *rp_m[2][6];
    for (int t = 0; t < 3; t++) wp_lin[t] = (__bf16*)take(WSZ * 2);
    for (int L = 0; L < 2; L++)
        for (int p = 0; p < 4; p++)
            for (int t = 0; t < 3; t++) wp_proj[L][p][t] = (__bf16*)take(WSZ * 2);
    for (int L = 0; L < 2; L++)
        for (int e = 0; e < 6; e++) {
            rp_a[L][e] = (__bf16*)take((size_t)NHEAD * 64 * 64 * 2);
            rp_m[L][e] = (__bf16*)take((size_t)NHEAD * 64 * 64 * 2);
        }

    auto cdiv = [](int a, int b) { return (a + b - 1) / b; };

    // ---- repack all weights to bf16 WMMA-B layout (deterministic, every call)
    const int RPT = 24 * 12 * 32;                        // threads per 384x384 matrix
    for (int t = 0; t < 3; t++)
        repack_b<<<dim3(cdiv(RPT, 256), 1), 256, 0, stream>>>(lin_w[t], wp_lin[t], DMODEL, DMODEL);
    for (int L = 0; L < 2; L++) {
        for (int p = 0; p < 4; p++)
            for (int t = 0; t < 3; t++)
                repack_b<<<dim3(cdiv(RPT, 256), 1), 256, 0, stream>>>(pw[L][p][t], wp_proj[L][p][t], DMODEL, DMODEL);
        for (int e = 0; e < 6; e++) {
            repack_b<<<dim3(1, NHEAD), 256, 0, stream>>>(arel[L][e], rp_a[L][e], 64, 64);
            repack_b<<<dim3(1, NHEAD), 256, 0, stream>>>(mrel[L][e], rp_m[L][e], 64, 64);
        }
    }

    // ---- encoder + input projection (ReLU)
    for (int t = 0; t < 3; t++) {
        encode_kernel<<<NN[t], 128, 0, stream>>>((const float*)d_in[t], gbuf);
        gemm384<1><<<dim3(NN[t] / 64, 6), 128, 0, stream>>>(
            gbuf, wp_lin[t], lin_b[t], x[t], nullptr, nullptr);
    }

    // ---- HGT layers
    for (int L = 0; L < 2; L++) {
        for (int t = 0; t < 3; t++) {
            gemm384<0><<<dim3(NN[t] / 64, 6), 128, 0, stream>>>(x[t], wp_proj[L][0][t], pb[L][0][t], kb[t], nullptr, nullptr);
            gemm384<0><<<dim3(NN[t] / 64, 6), 128, 0, stream>>>(x[t], wp_proj[L][1][t], pb[L][1][t], qb[t], nullptr, nullptr);
            gemm384<0><<<dim3(NN[t] / 64, 6), 128, 0, stream>>>(x[t], wp_proj[L][2][t], pb[L][2][t], vb[t], nullptr, nullptr);
            hipMemsetAsync(ob[t], 0, fb[t], stream);
        }
        for (int e = 0; e < 6; e++) {
            const int s = ETS[e].s, d = ETS[e].d, E = ETS[e].E;
            reltrans<<<dim3(NN[s] / 64, NHEAD, 2), 128, 0, stream>>>(
                kb[s], vb[s], rp_a[L][e], rp_m[L][e], krb, vrb);
            init_softmax<<<cdiv(NN[d] * NHEAD, 256), 256, 0, stream>>>(amaxB, denomB, NN[d] * NHEAD);
            alpha_kernel<<<cdiv(E * NHEAD, 256), 256, 0, stream>>>(
                qb[d], krb, ei[e], prel[L][e], alphaB, amaxB, E);
            expdenom_kernel<<<cdiv(E * NHEAD, 256), 256, 0, stream>>>(ei[e], alphaB, amaxB, denomB, E);
            scatter_kernel<<<cdiv(E * NHEAD * 16, 256), 256, 0, stream>>>(
                ei[e], alphaB, denomB, vrb, ob[d], E);
        }
        for (int t = 0; t < 3; t++) {
            gelu_kernel<<<cdiv(NN[t] * DMODEL, 256), 256, 0, stream>>>(ob[t], gbuf, NN[t] * DMODEL);
            gemm384<2><<<dim3(NN[t] / 64, 6), 128, 0, stream>>>(
                gbuf, wp_proj[L][3][t], pb[L][3][t], xn[t], skp[L][t], x[t]);
        }
        for (int t = 0; t < 3; t++) { float* tmp = x[t]; x[t] = xn[t]; xn[t] = tmp; }
    }

    hipMemcpyAsync(d_out, x[0], (size_t)NN[0] * DMODEL * 4, hipMemcpyDeviceToDevice, stream);
}
